// Model_49108656063207
// MI455X (gfx1250) — compile-verified, matrix operand order checked by reading, not run on hardware
//
#include <hip/hip_runtime.h>

// ---------------------------------------------------------------------------
// Model dims (from reference)
// ---------------------------------------------------------------------------
namespace {
constexpr int kB = 8, kS = 1024, kG = 4096;
constexpr int kD = 256, kH = 4, kL = 8;          // encoder
constexpr int kDD = 256, kHd = 2, kDH = 128, kLd = 4, kM = 256; // decoder
constexpr int kT = 100;
constexpr float kPAD = -100.0f;
}

typedef __attribute__((ext_vector_type(16))) __bf16 bf16x16;
typedef __attribute__((ext_vector_type(8)))  __bf16 bf16x8;
typedef __attribute__((ext_vector_type(8)))  float  f32x8;

// f32 -> bf16 via native hardware convert (v_cvt_*bf16* on gfx1250)
__device__ __forceinline__ __bf16 f2bf(float f) { return (__bf16)f; }

__device__ __forceinline__ float gelu_tanh(float x) {
  float x3 = x * x * x;
  return 0.5f * x * (1.0f + tanhf(0.7978845608028654f * (x + 0.044715f * x3)));
}

// ---------------------------------------------------------------------------
// Generic batched WMMA GEMM:  C[b] = epi(A[b] @ B[b] + bias) (+ Res[b])
// batch bz decomposes as (outer, inner) = (bz / nInner, bz % nInner).
// REQUIRES: M % 128 == 0, N % 64 == 0, K % 32 == 0 (true for every call here).
// Tile: 128(M) x 64(N) per workgroup; 8 waves, each wave computes 16 x 64 with
// one A fragment reused across four v_wmma_f32_16x16x32_bf16.
// Double-buffered LDS pipeline: global loads for tile s+1 are issued before the
// WMMAs of tile s, conversion+LDS stores after, one barrier per K-step.
// ---------------------------------------------------------------------------
struct GemmP {
  const float* A; int lda; long long aO, aI; int tA;
  const float* Bm; int ldb; long long bO, bI; int tB;
  const float* bias;
  const float* Res; int ldr; long long rO, rI;
  float* C; int ldc; long long cO, cI;
  int M, N, K, nInner, epi; // epi: 0 = none, 1 = gelu
};

__global__ __launch_bounds__(256) void gemm_wmma(GemmP p) {
  // 40-element row stride: 80 B (16 B aligned), stride of 20 banks ->
  // conflict-free b128 fragment loads across 16 lanes.
  __shared__ alignas(16) __bf16 As[2][128][40];
  __shared__ alignas(16) __bf16 BsT[2][64][40];   // B transposed: [n][k]

  const int bz = blockIdx.z;
  const int oi = bz / p.nInner, ii = bz % p.nInner;
  const float* __restrict__ A  = p.A  + oi * p.aO + ii * p.aI;
  const float* __restrict__ Bm = p.Bm + oi * p.bO + ii * p.bI;
  float*       __restrict__ C  = p.C  + oi * p.cO + ii * p.cI;
  const float* __restrict__ R  = p.Res ? (p.Res + oi * p.rO + ii * p.rI) : nullptr;

  const int m0 = blockIdx.y * 128, n0 = blockIdx.x * 64;
  const int tid = threadIdx.x;
  const int wave = tid >> 5, lane = tid & 31;
  const int wm = wave * 16;

  f32x8 acc[4];
#pragma unroll
  for (int j = 0; j < 4; ++j)
#pragma unroll
    for (int e = 0; e < 8; ++e) acc[j][e] = 0.f;

  // Per-lane fragment coordinates (loop-invariant).
  const int mr = wm + (lane & 15);
  const int kb = (lane >> 4) * 8;      // 0 or 8
  const int nc = lane & 15;

  // Staging decomposition (per thread).
  const int sKg  = tid & 3;            // k-group (8 consecutive k)
  const int sRow = tid >> 2;           // 0..63
  const int bK   = tid >> 3;           // 0..31  (non-transposed B path)
  const int bN   = (tid & 7) * 8;      // 0..56

  // ---- pipeline stage helpers ----
  float ra[2][8];   // A tile registers (two 64-row segments)
  float rb[8];      // B tile registers

  auto load_tile = [&](int k0) {
#pragma unroll
    for (int seg = 0; seg < 2; ++seg) {
      const int mrow = sRow + seg * 64;
      const int gk = k0 + sKg * 8;
      if (!p.tA) {
        const float4* s4 = (const float4*)(A + (long long)(m0 + mrow) * p.lda + gk);
        float4 u0 = s4[0], u1 = s4[1];
        ra[seg][0] = u0.x; ra[seg][1] = u0.y; ra[seg][2] = u0.z; ra[seg][3] = u0.w;
        ra[seg][4] = u1.x; ra[seg][5] = u1.y; ra[seg][6] = u1.z; ra[seg][7] = u1.w;
      } else {
        const float* s = A + (long long)gk * p.lda + (m0 + mrow);
#pragma unroll
        for (int j = 0; j < 8; ++j) ra[seg][j] = s[(long long)j * p.lda];
      }
    }
    if (p.tB) {
      const int gk = k0 + sKg * 8;
      const float4* s4 = (const float4*)(Bm + (long long)(n0 + sRow) * p.ldb + gk);
      float4 u0 = s4[0], u1 = s4[1];
      rb[0] = u0.x; rb[1] = u0.y; rb[2] = u0.z; rb[3] = u0.w;
      rb[4] = u1.x; rb[5] = u1.y; rb[6] = u1.z; rb[7] = u1.w;
    } else {
      const float4* s4 = (const float4*)(Bm + (long long)(k0 + bK) * p.ldb + (n0 + bN));
      float4 u0 = s4[0], u1 = s4[1];
      rb[0] = u0.x; rb[1] = u0.y; rb[2] = u0.z; rb[3] = u0.w;
      rb[4] = u1.x; rb[5] = u1.y; rb[6] = u1.z; rb[7] = u1.w;
    }
  };

  auto store_tile = [&](int buf) {
#pragma unroll
    for (int seg = 0; seg < 2; ++seg) {
      bf16x8 pk;
#pragma unroll
      for (int j = 0; j < 8; ++j) pk[j] = f2bf(ra[seg][j]);
      *(bf16x8*)&As[buf][sRow + seg * 64][sKg * 8] = pk;
    }
    if (p.tB) {
      bf16x8 pk;
#pragma unroll
      for (int j = 0; j < 8; ++j) pk[j] = f2bf(rb[j]);
      *(bf16x8*)&BsT[buf][sRow][sKg * 8] = pk;
    } else {
#pragma unroll
      for (int j = 0; j < 8; ++j) BsT[buf][bN + j][bK] = f2bf(rb[j]);
    }
  };

  const int nsteps = p.K >> 5;

  // prologue: stage tile 0
  load_tile(0);
  store_tile(0);
  __syncthreads();

  for (int s = 0; s < nsteps; ++s) {
    const int cur = s & 1;
    const bool more = (s + 1) < nsteps;

    // (1) issue global loads for the next tile (latency covered by WMMAs below)
    if (more) load_tile((s + 1) << 5);
    if (!p.tA && (s + 2) < nsteps && lane == 0)
      __builtin_prefetch(A + (long long)(m0 + wm) * p.lda + ((s + 2) << 5), 0, 1);

    // (2) preload ALL fragments from the current buffer (one DS clause),
    //     then run the 4 WMMAs so the waits are partial/overlapped.
    //     (ISA layout: frag i=0..7 -> K=kb..kb+7, i=8..15 -> K=kb+16..kb+23)
    bf16x8 al = *(const bf16x8*)&As[cur][mr][kb];
    bf16x8 ah = *(const bf16x8*)&As[cur][mr][kb + 16];
    bf16x16 af = __builtin_shufflevector(al, ah, 0, 1, 2, 3, 4, 5, 6, 7,
                                         8, 9, 10, 11, 12, 13, 14, 15);
    bf16x16 bfv[4];
#pragma unroll
    for (int j = 0; j < 4; ++j) {
      bf16x8 bl = *(const bf16x8*)&BsT[cur][j * 16 + nc][kb];
      bf16x8 bh = *(const bf16x8*)&BsT[cur][j * 16 + nc][kb + 16];
      bfv[j] = __builtin_shufflevector(bl, bh, 0, 1, 2, 3, 4, 5, 6, 7,
                                       8, 9, 10, 11, 12, 13, 14, 15);
    }
#pragma unroll
    for (int j = 0; j < 4; ++j)
      acc[j] = __builtin_amdgcn_wmma_f32_16x16x32_bf16(false, af, false, bfv[j],
                                                       (short)0, acc[j], false, false);

    // (3) convert + store next tile into the other buffer, (4) one barrier
    if (more) {
      store_tile(cur ^ 1);
      __syncthreads();
    }
  }

  // ---- epilogue + store (C/D layout: m = r + 8*(lane>>4), n = lane&15) ----
  const int half = lane >> 4;
#pragma unroll
  for (int j = 0; j < 4; ++j) {
    const int gn = n0 + j * 16 + nc;
    const float bia = p.bias ? p.bias[gn] : 0.f;
#pragma unroll
    for (int r = 0; r < 8; ++r) {
      const int gm = m0 + wm + r + 8 * half;
      float v = acc[j][r] + bia;
      if (p.epi == 1) v = gelu_tanh(v);
      if (R) v += R[(long long)gm * p.ldr + gn];
      C[(long long)gm * p.ldc + gn] = v;
    }
  }
}

// ---------------------------------------------------------------------------
// Embedding: bin softmax (T=100), exp_emb, sample_emb, x0, key mask bias
// ---------------------------------------------------------------------------
__global__ __launch_bounds__(128) void embed_k(
    const float* mtx, const int* samples, const float* alpha,
    const float* w1, const float* b1, const float* bin_emb, const float* table,
    float* out_bw, float* out_exp, float* out_samp, float* x0, float* keybias) {
  const int token = blockIdx.x;     // [0, B*S)
  const int tid = threadIdx.x;      // 128
  const float m = mtx[token];
  __shared__ float w[kT];
  __shared__ float red[128];

  float l = -3.4e38f;
  if (tid < kT) {
    float t = m * w1[tid] + b1[tid];
    l = (t > 0.f) ? t : 0.01f * t;  // leaky_relu, slope 0.01
  }
  red[tid] = l; __syncthreads();
  for (int s = 64; s > 0; s >>= 1) {
    if (tid < s) red[tid] = fmaxf(red[tid], red[tid + s]);
    __syncthreads();
  }
  float mx = red[0]; __syncthreads();
  float e = (tid < kT) ? __expf(l - mx) : 0.f;
  red[tid] = e; __syncthreads();
  for (int s = 64; s > 0; s >>= 1) {
    if (tid < s) red[tid] += red[tid + s];
    __syncthreads();
  }
  float inv = 1.f / red[0];
  if (tid < kT) {
    float bw = e * inv;
    w[tid] = bw;
    out_bw[(long long)token * kT + tid] = bw;
  }
  __syncthreads();

  const float a = alpha[0];
  const int samp = samples[token];
  for (int d = tid; d < kD; d += 128) {
    float acc = 0.f;
    for (int t = 0; t < kT; ++t) acc += w[t] * bin_emb[t * kD + d];
    float se = table[(long long)samp * kD + d];
    out_exp[(long long)token * kD + d] = acc;
    out_samp[(long long)token * kD + d] = se;
    x0[(long long)token * kD + d] = a * acc + se;
  }
  if (tid == 0) keybias[token] = (m != kPAD) ? 0.f : -1e9f;
}

// ---------------------------------------------------------------------------
// Row softmax over ncols with scale + per-key additive bias (pad mask)
// ---------------------------------------------------------------------------
__global__ __launch_bounds__(256) void softmax_rows_k(float* sc, const float* keybias,
                                                      int ncols, float scale) {
  const long long row = blockIdx.x;
  float* p = sc + row * ncols;
  __shared__ float red[256];
  const int tid = threadIdx.x;

  float mx = -3.4e38f;
  for (int c = tid; c < ncols; c += 256) {
    float v = p[c] * scale + keybias[c];
    p[c] = v;
    mx = fmaxf(mx, v);
  }
  red[tid] = mx; __syncthreads();
  for (int s = 128; s > 0; s >>= 1) {
    if (tid < s) red[tid] = fmaxf(red[tid], red[tid + s]);
    __syncthreads();
  }
  mx = red[0]; __syncthreads();
  float sum = 0.f;
  for (int c = tid; c < ncols; c += 256) {
    float e = __expf(p[c] - mx);
    p[c] = e;
    sum += e;
  }
  red[tid] = sum; __syncthreads();
  for (int s = 128; s > 0; s >>= 1) {
    if (tid < s) red[tid] += red[tid + s];
    __syncthreads();
  }
  float inv = 1.f / red[0];
  for (int c = tid; c < ncols; c += 256) p[c] *= inv;
}

// ---------------------------------------------------------------------------
// LayerNorm over last dim (256); in may equal out
// ---------------------------------------------------------------------------
__global__ __launch_bounds__(256) void layernorm_k(const float* in, float* out,
                                                   const float* g, const float* b) {
  const long long row = blockIdx.x;
  const int d = threadIdx.x;
  __shared__ float red[256];
  float v = in[row * 256 + d];
  red[d] = v; __syncthreads();
  for (int s = 128; s > 0; s >>= 1) {
    if (d < s) red[d] += red[d + s];
    __syncthreads();
  }
  float mean = red[0] * (1.f / 256.f); __syncthreads();
  float c = v - mean;
  red[d] = c * c; __syncthreads();
  for (int s = 128; s > 0; s >>= 1) {
    if (d < s) red[d] += red[d + s];
    __syncthreads();
  }
  float var = red[0] * (1.f / 256.f);
  out[row * 256 + d] = c * rsqrtf(var + 1e-5f) * g[d] + b[d];
}

// encoder head merge: [B,H,S,64] -> [B,S,256]
__global__ __launch_bounds__(256) void enc_merge_k(const float* in, float* out) {
  long long i = (long long)blockIdx.x * 256 + threadIdx.x;
  if (i >= (long long)kB * kH * kS * 64) return;
  int d = (int)(i & 63);
  long long t = i >> 6;
  int s = (int)(t % kS); t /= kS;
  int h = (int)(t % kH);
  int b = (int)(t / kH);
  out[((long long)(b * kS + s)) * kD + h * 64 + d] = in[i];
}

// extend: fill base values (masked > nan > zero)
__global__ __launch_bounds__(256) void ext_base_k(const unsigned char* nans,
                                                  const unsigned char* masked,
                                                  const float* filt, float* ext) {
  long long i = (long long)blockIdx.x * 256 + threadIdx.x;
  if (i >= (long long)kB * kG * kD) return;
  int d = (int)(i % kD);
  long long bg = i / kD;
  float v = 0.f;
  if (nans[bg]) v = filt[d];
  if (masked[bg]) v = filt[kD + d];
  ext[i] = v;
}

// extend: scatter encoder outputs to raw positions not overwritten by filt
__global__ __launch_bounds__(256) void ext_scatter_k(const int* map_cols,
                                                     const unsigned char* nans,
                                                     const unsigned char* masked,
                                                     const float* x, float* ext) {
  const int token = blockIdx.x;  // [0, B*S)
  const int d = threadIdx.x;     // 256
  const int b = token / kS;
  const int g = map_cols[token];
  const long long bg = (long long)b * kG + g;
  if (nans[bg] || masked[bg]) return;
  ext[bg * kD + d] = x[(long long)token * kD + d];
}

// FAVOR+ phi: in-place on u[Hd,G,256]; sect 0 = q, 1 = k (within qkv row)
__global__ __launch_bounds__(256) void phi_k(const float* qkvb, int sect, float* u) {
  const int idx = blockIdx.x;          // [0, Hd*G)
  const int n = idx % kG, h = idx / kG;
  const int tid = threadIdx.x;         // 256
  __shared__ float red[256];
  const float sc = 0.29730177875068026f;  // 128^-0.25

  const float* t = qkvb + (long long)n * (3 * kDD) + sect * kDD + h * kDH;
  float sq = 0.f;
  if (tid < kDH) { float q = t[tid] * sc; sq = q * q; }
  red[tid] = sq; __syncthreads();
  for (int s = 128; s > 0; s >>= 1) {
    if (tid < s) red[tid] += red[tid + s];
    __syncthreads();
  }
  float qn = red[0]; __syncthreads();

  float* urow = u + (long long)idx * kM;
  float uval = urow[tid] * sc;
  red[tid] = uval; __syncthreads();
  for (int s = 128; s > 0; s >>= 1) {
    if (tid < s) red[tid] = fmaxf(red[tid], red[tid + s]);
    __syncthreads();
  }
  float um = red[0];
  urow[tid] = __expf(uval - 0.5f * qn - um) * (1.0f / 16.0f) + 1e-6f; // 1/sqrt(256)
}

// ksum[h,m] = sum_n kp[h,n,m]
__global__ __launch_bounds__(256) void ksum_k(const float* kp, float* ks) {
  const int h = blockIdx.x, m = threadIdx.x;
  const float* base = kp + (long long)h * kG * kM;
  float s = 0.f;
  for (int n = 0; n < kG; ++n) s += base[(long long)n * kM + m];
  ks[h * kM + m] = s;
}

// z[h,n] = 1 / (qp[h,n,:] . ksum[h,:] + 1e-6)
__global__ __launch_bounds__(256) void zinv_k(const float* qp, const float* ks, float* z) {
  int i = blockIdx.x * 256 + threadIdx.x;  // [0, Hd*G)
  if (i >= kHd * kG) return;
  int h = i / kG;
  const float* row = qp + (long long)i * kM;
  const float* kk = ks + h * kM;
  float s = 0.f;
  for (int m = 0; m < kM; ++m) s += row[m] * kk[m];
  z[i] = 1.f / (s + 1e-6f);
}

// decoder merge+scale: [Hd,G,128]*z -> ctx_b[n, h*128+d]
__global__ __launch_bounds__(256) void dec_merge_k(const float* atd, const float* zb,
                                                   float* ctx_b) {
  long long i = (long long)blockIdx.x * 256 + threadIdx.x;
  if (i >= (long long)kHd * kG * kDH) return;
  int d = (int)(i % kDH);
  long long t = i / kDH;
  int n = (int)(t % kG);
  int h = (int)(t / kG);
  ctx_b[(long long)n * kDD + h * kDH + d] = atd[i] * zb[h * kG + n];
}

// ---------------------------------------------------------------------------
// Host-side launch helper
// ---------------------------------------------------------------------------
static inline void launch_gemm(hipStream_t st,
    const float* A, int lda, long long aO, long long aI, int tA,
    const float* Bm, int ldb, long long bO, long long bI, int tB,
    const float* bias,
    const float* Res, int ldr, long long rO, long long rI,
    float* C, int ldc, long long cO, long long cI,
    int M, int N, int K, int batches, int nInner, int epi) {
  GemmP p{A, lda, aO, aI, tA, Bm, ldb, bO, bI, tB, bias,
          Res, ldr, rO, rI, C, ldc, cO, cI, M, N, K, nInner, epi};
  dim3 grid((N + 63) / 64, (M + 127) / 128, batches);
  gemm_wmma<<<grid, dim3(256), 0, st>>>(p);
}

extern "C" void kernel_launch(void* const* d_in, const int* in_sizes, int n_in,
                              void* d_out, int out_size, void* d_ws, size_t ws_size,
                              hipStream_t stream) {
  (void)in_sizes; (void)out_size; (void)ws_size;
  if (n_in < 38) return;

  const float* mtx      = (const float*)d_in[0];
  const int*   samples  = (const int*)d_in[1];
  const int*   map_cols = (const int*)d_in[2];
  const unsigned char* nans   = (const unsigned char*)d_in[3];
  const unsigned char* masked = (const unsigned char*)d_in[4];
  const float* alpha   = (const float*)d_in[5];
  const float* bin_w1  = (const float*)d_in[6];
  const float* bin_b1  = (const float*)d_in[7];
  const float* bin_emb = (const float*)d_in[8];
  const float* table   = (const float*)d_in[9];
  const float* filt    = (const float*)d_in[10];
  const float* eWqkv = (const float*)d_in[11]; const float* ebqkv = (const float*)d_in[12];
  const float* eWo   = (const float*)d_in[13]; const float* ebo   = (const float*)d_in[14];
  const float* eg1   = (const float*)d_in[15]; const float* ebb1  = (const float*)d_in[16];
  const float* eW1   = (const float*)d_in[17]; const float* ef1   = (const float*)d_in[18];
  const float* eW2   = (const float*)d_in[19]; const float* ef2   = (const float*)d_in[20];
  const float* eg2   = (const float*)d_in[21]; const float* ebb2  = (const float*)d_in[22];
  const float* fcw   = (const float*)d_in[23]; const float* fcb   = (const float*)d_in[24];
  const float* dWqkv = (const float*)d_in[25]; const float* dbqkv = (const float*)d_in[26];
  const float* dWo   = (const float*)d_in[27]; const float* dbo   = (const float*)d_in[28];
  const float* dg1   = (const float*)d_in[29]; const float* dbb1  = (const float*)d_in[30];
  const float* dW1   = (const float*)d_in[31]; const float* df1   = (const float*)d_in[32];
  const float* dW2   = (const float*)d_in[33]; const float* df2   = (const float*)d_in[34];
  const float* dg2   = (const float*)d_in[35]; const float* dbb2  = (const float*)d_in[36];
  const float* dproj = (const float*)d_in[37];

  float* out = (float*)d_out;
  float* y     = out;                                     // [B,G,256]
  float* oExp  = out + (size_t)kB * kG * kDD;             // [B,S,256]
  float* oBW   = oExp + (size_t)kB * kS * kD;             // [B,S,100]
  float* oSamp = oBW + (size_t)kB * kS * kT;              // [B,S,256]

  // ---- workspace layout (phases reused; peak ~135 MB) ----
  float* W = (float*)d_ws;
  // encoder phase
  float* x_e   = W;
  float* keyb  = x_e   + (size_t)kB * kS * kD;
  float* qkv_e = keyb  + (size_t)kB * kS;
  float* sc_e  = qkv_e + (size_t)kB * kS * 3 * kD;        // per-b scores [H,S,S]
  float* at_e  = sc_e  + (size_t)kH * kS * kS;
  float* cx_e  = at_e  + (size_t)kB * kH * kS * 64;
  float* ff_e  = cx_e  + (size_t)kB * kS * kD;
  float* ext   = ff_e  + (size_t)kB * kS * 4 * kD;

  // ---- embedding (also writes exp_emb / bin_weights / sample_emb outputs) ----
  embed_k<<<kB * kS, 128, 0, stream>>>(mtx, samples, alpha, bin_w1, bin_b1,
                                       bin_emb, table, oBW, oExp, oSamp, x_e, keyb);

  const float iscale = 0.125f;  // 1/sqrt(dh=64)
  for (int l = 0; l < kL; ++l) {
    // qkv = x @ Wqkv + b   [8192 x 768]
    launch_gemm(stream, x_e, kD, 0, 0, 0,
                eWqkv + (size_t)l * kD * 3 * kD, 3 * kD, 0, 0, 0,
                ebqkv + (size_t)l * 3 * kD,
                nullptr, 0, 0, 0,
                qkv_e, 3 * kD, 0, 0, kB * kS, 3 * kD, kD, 1, 1, 0);
    for (int b = 0; b < kB; ++b) {
      const float* qb = qkv_e + (size_t)b * kS * 3 * kD;
      // scores[h] = Q_h @ K_h^T   (M=N=S, K=64), batched over heads
      launch_gemm(stream, qb, 3 * kD, 0, 64, 0,
                  qb + kD, 3 * kD, 0, 64, 1,
                  nullptr, nullptr, 0, 0, 0,
                  sc_e, kS, 0, (long long)kS * kS,
                  kS, kS, 64, kH, kH, 0);
      softmax_rows_k<<<kH * kS, 256, 0, stream>>>(sc_e, keyb + (size_t)b * kS, kS, iscale);
      // ctx[h] = P_h @ V_h  (M=S, N=64, K=S)
      launch_gemm(stream, sc_e, kS, 0, (long long)kS * kS, 0,
                  qb + 2 * kD, 3 * kD, 0, 64, 0,
                  nullptr, nullptr, 0, 0, 0,
                  at_e + (size_t)b * kH * kS * 64, 64, 0, (long long)kS * 64,
                  kS, 64, kS, kH, kH, 0);
    }
    {
      long long tot = (long long)kB * kH * kS * 64;
      enc_merge_k<<<(unsigned)((tot + 255) / 256), 256, 0, stream>>>(at_e, cx_e);
    }
    // x = LN(x + ctx @ Wo + bo)
    launch_gemm(stream, cx_e, kD, 0, 0, 0,
                eWo + (size_t)l * kD * kD, kD, 0, 0, 0,
                ebo + (size_t)l * kD,
                x_e, kD, 0, 0,
                x_e, kD, 0, 0, kB * kS, kD, kD, 1, 1, 0);
    layernorm_k<<<kB * kS, 256, 0, stream>>>(x_e, x_e, eg1 + (size_t)l * kD, ebb1 + (size_t)l * kD);
    // x = LN(x + gelu(x@W1+b1)@W2 + b2)
    launch_gemm(stream, x_e, kD, 0, 0, 0,
                eW1 + (size_t)l * kD * 4 * kD, 4 * kD, 0, 0, 0,
                ef1 + (size_t)l * 4 * kD,
                nullptr, 0, 0, 0,
                ff_e, 4 * kD, 0, 0, kB * kS, 4 * kD, kD, 1, 1, 1);
    launch_gemm(stream, ff_e, 4 * kD, 0, 0, 0,
                eW2 + (size_t)l * 4 * kD * kD, kD, 0, 0, 0,
                ef2 + (size_t)l * kD,
                x_e, kD, 0, 0,
                x_e, kD, 0, 0, kB * kS, kD, 4 * kD, 1, 1, 0);
    layernorm_k<<<kB * kS, 256, 0, stream>>>(x_e, x_e, eg2 + (size_t)l * kD, ebb2 + (size_t)l * kD);
  }

  // ---- extend + fc ----
  {
    long long tot = (long long)kB * kG * kD;
    ext_base_k<<<(unsigned)((tot + 255) / 256), 256, 0, stream>>>(nans, masked, filt, ext);
    ext_scatter_k<<<kB * kS, kD, 0, stream>>>(map_cols, nans, masked, x_e, ext);
  }
  launch_gemm(stream, ext, kD, 0, 0, 0,
              fcw, kDD, 0, 0, 0, fcb,
              nullptr, 0, 0, 0,
              y, kDD, 0, 0, kB * kG, kDD, kD, 1, 1, 0);

  // ---- decoder (Performer / FAVOR+), workspace reused from offset 0 ----
  float* hdec = W;
  float* qkvb = hdec + (size_t)kB * kG * kDD;          // per-b [G,768]
  float* uq   = qkvb + (size_t)kG * 3 * kDD;           // per-b [Hd,G,256]
  float* uk   = uq   + (size_t)kHd * kG * kM;
  float* kvb  = uk   + (size_t)kHd * kG * kM;          // [Hd,256,128]
  float* ksum = kvb  + (size_t)kHd * kM * kDH;         // [Hd,256]
  float* zb   = ksum + (size_t)kHd * kM;               // [Hd,G]
  float* atd  = zb   + (size_t)kHd * kG;               // [Hd,G,128]
  float* cxd  = atd  + (size_t)kHd * kG * kDH;         // [B,G,256]
  float* ffd  = cxd  + (size_t)kB * kG * kDD;          // [8192,1024]

  for (int l = 0; l < kLd; ++l) {
    layernorm_k<<<kB * kG, 256, 0, stream>>>(y, hdec, dg1 + (size_t)l * kDD, dbb1 + (size_t)l * kDD);
    for (int b = 0; b < kB; ++b) {
      launch_gemm(stream, hdec + (size_t)b * kG * kDD, kDD, 0, 0, 0,
                  dWqkv + (size_t)l * kDD * 3 * kDD, 3 * kDD, 0, 0, 0,
                  dbqkv + (size_t)l * 3 * kDD,
                  nullptr, 0, 0, 0,
                  qkvb, 3 * kDD, 0, 0, kG, 3 * kDD, kDD, 1, 1, 0);
      // u = t @ proj^T  (M=G, N=256, K=128), batched over heads
      launch_gemm(stream, qkvb, 3 * kDD, 0, kDH, 0,
                  dproj, kDH, 0, 0, 1,
                  nullptr, nullptr, 0, 0, 0,
                  uq, kM, 0, (long long)kG * kM, kG, kM, kDH, kHd, kHd, 0);
      launch_gemm(stream, qkvb + kDD, 3 * kDD, 0, kDH, 0,
                  dproj, kDH, 0, 0, 1,
                  nullptr, nullptr, 0, 0, 0,
                  uk, kM, 0, (long long)kG * kM, kG, kM, kDH, kHd, kHd, 0);
      phi_k<<<kHd * kG, 256, 0, stream>>>(qkvb, 0, uq);
      phi_k<<<kHd * kG, 256, 0, stream>>>(qkvb, 1, uk);
      // kv[h] = kp^T @ V  (M=256, N=128, K=G)
      launch_gemm(stream, uk, kM, 0, (long long)kG * kM, 1,
                  qkvb + 2 * kDD, 3 * kDD, 0, kDH, 0,
                  nullptr, nullptr, 0, 0, 0,
                  kvb, kDH, 0, (long long)kM * kDH, kM, kDH, kG, kHd, kHd, 0);
      ksum_k<<<kHd, kM, 0, stream>>>(uk, ksum);
      zinv_k<<<(kHd * kG + 255) / 256, 256, 0, stream>>>(uq, ksum, zb);
      // attn[h] = qp @ kv  (M=G, N=128, K=256)
      launch_gemm(stream, uq, kM, 0, (long long)kG * kM, 0,
                  kvb, kDH, 0, (long long)kM * kDH, 0,
                  nullptr, nullptr, 0, 0, 0,
                  atd, kDH, 0, (long long)kG * kDH, kG, kDH, kM, kHd, kHd, 0);
      {
        long long tot = (long long)kHd * kG * kDH;
        dec_merge_k<<<(unsigned)((tot + 255) / 256), 256, 0, stream>>>(
            atd, zb, cxd + (size_t)b * kG * kDD);
      }
    }
    // y = y + ctx @ Wo + bo
    launch_gemm(stream, cxd, kDD, 0, 0, 0,
                dWo + (size_t)l * kDD * kDD, kDD, 0, 0, 0,
                dbo + (size_t)l * kDD,
                y, kDD, 0, 0,
                y, kDD, 0, 0, kB * kG, kDD, kDD, 1, 1, 0);
    // y = y + gelu(LN(y)@W1+b1)@W2 + b2   (FFN chunked over 4 x 8192 rows)
    layernorm_k<<<kB * kG, 256, 0, stream>>>(y, hdec, dg2 + (size_t)l * kDD, dbb2 + (size_t)l * kDD);
    for (int c = 0; c < 4; ++c) {
      float* hc = hdec + (size_t)c * 8192 * kDD;
      float* yc = y + (size_t)c * 8192 * kDD;
      launch_gemm(stream, hc, kDD, 0, 0, 0,
                  dW1 + (size_t)l * kDD * 4 * kDD, 4 * kDD, 0, 0, 0,
                  df1 + (size_t)l * 4 * kDD,
                  nullptr, 0, 0, 0,
                  ffd, 4 * kDD, 0, 0, 8192, 4 * kDD, kDD, 1, 1, 1);
      launch_gemm(stream, ffd, 4 * kDD, 0, 0, 0,
                  dW2 + (size_t)l * 4 * kDD * kDD, kDD, 0, 0, 0,
                  df2 + (size_t)l * kDD,
                  yc, kDD, 0, 0,
                  yc, kDD, 0, 0, 8192, kDD, 4 * kDD, 1, 1, 0);
    }
  }
}
